// FlaxLlamaAttention_15710990369538
// MI455X (gfx1250) — compile-verified, matrix operand order checked by reading, not run on hardware
//
#include <hip/hip_runtime.h>
#include <hip/hip_bf16.h>

typedef _Float16 v16h __attribute__((ext_vector_type(16)));
typedef float    v8f  __attribute__((ext_vector_type(8)));
typedef unsigned u32x4 __attribute__((ext_vector_type(4)));
typedef int      i32x4 __attribute__((ext_vector_type(4)));
typedef int      i32x8 __attribute__((ext_vector_type(8)));

union FragH { v16h h; unsigned u[8]; };
union HV8  { uint4 q; _Float16 h[8]; };

// K-offset of the half-pair held in fragment VGPR j for this lane-half,
// per CDNA5 ISA 16-bit A-matrix 16x32 layout (7.12.2).
__device__ __forceinline__ int fragBase(int j, int half) {
    return ((j & 4) ? 16 : 0) + ((j & 3) << 1) + (half << 3);
}

// ---- Tensor Data Mover: 2D tile (16-bit elems) global -> LDS, with LDS row pad.
// tile_d0 elems per line, tile_d1 lines, global line stride strideElems.
// padIntCode: pad after 2^(code+1) DWORDs; padAmtCode: pad (code+1) DWORDs.
__device__ __forceinline__ void tdm_load_2d(unsigned ldsOff, const _Float16* gptr,
                                            int tile_d0, int tile_d1,
                                            unsigned strideElems,
                                            int padIntCode, int padAmtCode) {
    unsigned long long ga = (unsigned long long)(size_t)(const void*)gptr;
    u32x4 g0;
    g0[0] = 1u;                                            // count=1 (valid user D#)
    g0[1] = ldsOff;                                        // lds_addr
    g0[2] = (unsigned)(ga & 0xFFFFFFFFull);                // global_addr[31:0]
    g0[3] = (unsigned)((ga >> 32) & 0x1FFFFFFull) | (2u << 30);  // addr[56:32], type=2
    unsigned td0 = strideElems;        // tensor_dim0 (generous; no OOB clip)
    unsigned td1 = 1u << 22;           // tensor_dim1 (generous)
    i32x8 g1;
    g1[0] = (int)((1u << 16)                               // data_size = 2 bytes
                | (1u << 20)                               // pad_enable
                | ((unsigned)padIntCode << 22)
                | ((unsigned)padAmtCode << 25));
    g1[1] = (int)((td0 & 0xFFFFu) << 16);                  // atomic_addr=0 | dim0 lo16
    g1[2] = (int)(((td0 >> 16) & 0xFFFFu) | ((td1 & 0xFFFFu) << 16));
    g1[3] = (int)(((td1 >> 16) & 0xFFFFu) | ((unsigned)tile_d0 << 16));
    g1[4] = (int)((unsigned)tile_d1 & 0xFFFFu);            // tile_dim1 | tile_dim2=0
    g1[5] = (int)strideElems;                              // tensor_dim0_stride[31:0]
    g1[6] = 0;
    g1[7] = 0;
    i32x4 gz = {0, 0, 0, 0};
#if defined(__clang_major__) && (__clang_major__ >= 23)
    i32x8 gz8 = {0, 0, 0, 0, 0, 0, 0, 0};
    __builtin_amdgcn_tensor_load_to_lds(g0, g1, gz, gz, gz8, 0);
#else
    __builtin_amdgcn_tensor_load_to_lds(g0, g1, gz, gz, 0);
#endif
}

// ---------------- f32 -> f16 conversion ----------------
__global__ void cvt_f32_f16(const float* __restrict__ src, _Float16* __restrict__ dst, int n) {
    for (int i = blockIdx.x * blockDim.x + threadIdx.x; i < n; i += gridDim.x * blockDim.x)
        dst[i] = (_Float16)src[i];
}

// ---------------- RoPE (in place on Q and K, f16) ----------------
__global__ void rope_kernel(_Float16* __restrict__ Q, _Float16* __restrict__ K,
                            const int* __restrict__ pos_ids, int total) {
    int id = blockIdx.x * blockDim.x + threadIdx.x;
    if (id >= total) return;
    int j   = id & 63;
    int t   = id >> 6;
    int h   = t & 15;
    int row = t >> 4;                 // 0..4095 == b*S+s
    float pos = (float)pos_ids[row];
    float ang = pos * __expf(((float)j) * (-9.210340371976184f / 64.0f));
    float s, c;
    __sincosf(ang, &s, &c);
    size_t base = (size_t)row * 2048 + h * 128 + j;
    float q1 = (float)Q[base], q2 = (float)Q[base + 64];
    Q[base]      = (_Float16)(q1 * c - q2 * s);
    Q[base + 64] = (_Float16)(q2 * c + q1 * s);
    float k1 = (float)K[base], k2 = (float)K[base + 64];
    K[base]      = (_Float16)(k1 * c - k2 * s);
    K[base + 64] = (_Float16)(k2 * c + k1 * s);
}

// ---------------- WMMA GEMM: C[M,N] = A[M,K] * B[K,N] ----------------
// block tile 128x128, 8 waves, wave tile 32x64, K chunk 32,
// double-buffered LDS: TDM (A) + wide loads (B) for tile t+1 overlap WMMAs of tile t.
__launch_bounds__(256)
__global__ void gemm_f16_wmma(const _Float16* __restrict__ A,
                              const _Float16* __restrict__ Bm,
                              _Float16* __restrict__ outH,
                              float* __restrict__ outF,
                              int M, int N, int K) {
    __shared__ _Float16 As[2][128][40];   // TDM filled (32 K + pad 8)
    __shared__ _Float16 Bt[2][128][40];   // transposed B (cols x K)

    const int tid  = threadIdx.x;
    const int w    = tid >> 5, lane = tid & 31;
    const int half = lane >> 4, ln = lane & 15;
    const int wr   = (w & 3) * 32;
    const int wc   = (w >> 2) * 64;
    const int m0   = blockIdx.y * 128, n0 = blockIdx.x * 128;
    const unsigned AsOff0 = (unsigned)(size_t)(void*)&As[0][0][0];
    const unsigned AsOff1 = (unsigned)(size_t)(void*)&As[1][0][0];

    // per-thread B-staging coordinates (2 chunks of 8 halves)
    int bk[2], bc[2];
#pragma unroll
    for (int it = 0; it < 2; ++it) {
        int idx = tid + it * 256;          // 0..511
        bk[it] = idx >> 4;                 // k row 0..31
        bc[it] = (idx & 15) << 3;          // col start 0..120
    }

    const v8f zero8 = {0.f,0.f,0.f,0.f,0.f,0.f,0.f,0.f};
    v8f acc[2][4];
#pragma unroll
    for (int mi = 0; mi < 2; ++mi)
#pragma unroll
        for (int ni = 0; ni < 4; ++ni) acc[mi][ni] = zero8;

    const int nK = K >> 5;

    // prologue: stage tile 0 into buffer 0
    if (w == 0)
        tdm_load_2d(AsOff0, A + (size_t)m0 * K, 32, 128, (unsigned)K, 3, 3);
    {
        HV8 u0, u1;
        u0.q = *(const uint4*)&Bm[(size_t)bk[0] * N + n0 + bc[0]];
        u1.q = *(const uint4*)&Bm[(size_t)bk[1] * N + n0 + bc[1]];
#pragma unroll
        for (int e = 0; e < 8; ++e) Bt[0][bc[0] + e][bk[0]] = u0.h[e];
#pragma unroll
        for (int e = 0; e < 8; ++e) Bt[0][bc[1] + e][bk[1]] = u1.h[e];
    }

    for (int t = 0; t < nK; ++t) {
        const int cur = t & 1, nxt = cur ^ 1;
        if (w == 0) __builtin_amdgcn_s_wait_tensorcnt(0);   // tile t DMA done
        __syncthreads();                                    // + all done with buf nxt

        const bool more = (t + 1 < nK);
        HV8 u0, u1;
        if (more) {
            // issue next tile's DMA + global loads; latency hidden by WMMAs below
            if (w == 0)
                tdm_load_2d(nxt ? AsOff1 : AsOff0,
                            A + (size_t)m0 * K + (t + 1) * 32, 32, 128,
                            (unsigned)K, 3, 3);
            u0.q = *(const uint4*)&Bm[(size_t)((t + 1) * 32 + bk[0]) * N + n0 + bc[0]];
            u1.q = *(const uint4*)&Bm[(size_t)((t + 1) * 32 + bk[1]) * N + n0 + bc[1]];
        }

        FragH af[2], bf[4];
#pragma unroll
        for (int mi = 0; mi < 2; ++mi) {
            int m = wr + mi * 16 + ln;
#pragma unroll
            for (int j = 0; j < 8; ++j)
                af[mi].u[j] = *(const unsigned*)&As[cur][m][fragBase(j, half)];
        }
#pragma unroll
        for (int ni = 0; ni < 4; ++ni) {
            int n = wc + ni * 16 + ln;
#pragma unroll
            for (int j = 0; j < 8; ++j)
                bf[ni].u[j] = *(const unsigned*)&Bt[cur][n][fragBase(j, half)];
        }
#pragma unroll
        for (int mi = 0; mi < 2; ++mi)
#pragma unroll
            for (int ni = 0; ni < 4; ++ni)
                acc[mi][ni] = __builtin_amdgcn_wmma_f32_16x16x32_f16(
                    false, af[mi].h, false, bf[ni].h, (short)0, acc[mi][ni], false, false);

        if (more) {
#pragma unroll
            for (int e = 0; e < 8; ++e) Bt[nxt][bc[0] + e][bk[0]] = u0.h[e];
#pragma unroll
            for (int e = 0; e < 8; ++e) Bt[nxt][bc[1] + e][bk[1]] = u1.h[e];
        }
    }

#pragma unroll
    for (int mi = 0; mi < 2; ++mi)
#pragma unroll
        for (int ni = 0; ni < 4; ++ni)
#pragma unroll
            for (int i = 0; i < 8; ++i) {
                int rg = m0 + wr + mi * 16 + i + half * 8;
                int cg = n0 + wc + ni * 16 + ln;
                float v = acc[mi][ni][i];
                if (outH) outH[(size_t)rg * N + cg] = (_Float16)v;
                else      outF[(size_t)rg * N + cg] = v;
            }
}

// ---------------- Flash attention (causal), f16 in / f16 out ----------------
__launch_bounds__(256)
__global__ void attn_kernel(const _Float16* __restrict__ Q,
                            const _Float16* __restrict__ Kx,
                            const _Float16* __restrict__ V,
                            _Float16* __restrict__ O) {
    __shared__ _Float16 Qs[128][136];   // TDM filled (128 + pad 8)
    __shared__ _Float16 Ks[32][136];    // TDM filled
    __shared__ _Float16 Vt[128][40];    // d x 32 keys (transposed)
    __shared__ _Float16 Ps[8][16][40];  // per-wave P tile

    const int tid  = threadIdx.x;
    const int w    = tid >> 5, lane = tid & 31;
    const int half = lane >> 4, ln = lane & 15;
    const int qb   = blockIdx.x;
    const int b    = blockIdx.y >> 4, h = blockIdx.y & 15;
    const float scale = 0.08838834764831845f;   // 1/sqrt(128)
    const unsigned QsOff = (unsigned)(size_t)(void*)&Qs[0][0];
    const unsigned KsOff = (unsigned)(size_t)(void*)&Ks[0][0];

    // Q block 128x128 via TDM
    if (w == 0) {
        tdm_load_2d(QsOff, Q + (size_t)(b * 2048 + qb * 128) * 2048 + h * 128,
                    128, 128, 2048u, 5, 3);
        __builtin_amdgcn_s_wait_tensorcnt(0);
    }
    __syncthreads();

    FragH qf[4];
    {
        int m = w * 16 + ln;
#pragma unroll
        for (int kk = 0; kk < 4; ++kk)
#pragma unroll
            for (int j = 0; j < 8; ++j)
                qf[kk].u[j] = *(const unsigned*)&Qs[m][kk * 32 + fragBase(j, half)];
    }

    const v8f zero8 = {0.f,0.f,0.f,0.f,0.f,0.f,0.f,0.f};
    v8f Oacc[8];
#pragma unroll
    for (int t = 0; t < 8; ++t) Oacc[t] = zero8;
    float m8[8], l8[8];
#pragma unroll
    for (int i = 0; i < 8; ++i) { m8[i] = -3.0e38f; l8[i] = 0.0f; }

    // per-thread V-staging coordinates
    int vk[2], vc[2];
#pragma unroll
    for (int it = 0; it < 2; ++it) {
        int idx = tid + it * 256;
        vk[it] = idx >> 4;                 // key 0..31
        vc[it] = (idx & 15) << 3;          // d start 0..120
    }

    const int nkt = (qb + 1) * 4;   // causal bound
    for (int kt = 0; kt < nkt; ++kt) {
        __syncthreads();            // previous Ks/Vt reads done
        if (w == 0)
            tdm_load_2d(KsOff, Kx + (size_t)(b * 2048 + kt * 32) * 2048 + h * 128,
                        128, 32, 2048u, 5, 3);
        // issue V loads now; scatter after the score WMMAs (latency hiding)
        HV8 uv0, uv1;
        uv0.q = *(const uint4*)&V[(size_t)(b * 2048 + kt * 32 + vk[0]) * 2048 + h * 128 + vc[0]];
        uv1.q = *(const uint4*)&V[(size_t)(b * 2048 + kt * 32 + vk[1]) * 2048 + h * 128 + vc[1]];
        if (w == 0) __builtin_amdgcn_s_wait_tensorcnt(0);
        __syncthreads();            // Ks ready

        // scores: 16 q rows x 32 keys, accumulate over d=128
        v8f sacc[2] = { zero8, zero8 };
#pragma unroll
        for (int c = 0; c < 2; ++c) {
            int key = c * 16 + ln;
#pragma unroll
            for (int kk = 0; kk < 4; ++kk) {
                FragH bf;
#pragma unroll
                for (int j = 0; j < 8; ++j)
                    bf.u[j] = *(const unsigned*)&Ks[key][kk * 32 + fragBase(j, half)];
                sacc[c] = __builtin_amdgcn_wmma_f32_16x16x32_f16(
                    false, qf[kk].h, false, bf.h, (short)0, sacc[c], false, false);
            }
        }

        // scatter V into Vt (needed only after the next barrier)
#pragma unroll
        for (int e = 0; e < 8; ++e) Vt[vc[0] + e][vk[0]] = uv0.h[e];
#pragma unroll
        for (int e = 0; e < 8; ++e) Vt[vc[1] + e][vk[1]] = uv1.h[e];

        // scale + causal mask + online softmax
        float p[2][8], alpha[8];
#pragma unroll
        for (int i = 0; i < 8; ++i) {
            int qg = qb * 128 + w * 16 + i + half * 8;
            float sv[2], mx = -3.0e38f;
#pragma unroll
            for (int c = 0; c < 2; ++c) {
                int kg = kt * 32 + c * 16 + ln;
                float s = sacc[c][i] * scale;
                if (kg > qg) s = -1.0e30f;
                sv[c] = s;
                mx = fmaxf(mx, s);
            }
#pragma unroll
            for (int off = 1; off < 16; off <<= 1)
                mx = fmaxf(mx, __shfl_xor(mx, off, 32));
            float mnew = fmaxf(m8[i], mx);
            alpha[i] = __expf(m8[i] - mnew);
            float rs = 0.f;
#pragma unroll
            for (int c = 0; c < 2; ++c) {
                float pe = __expf(sv[c] - mnew);
                p[c][i] = pe;
                rs += pe;
            }
#pragma unroll
            for (int off = 1; off < 16; off <<= 1)
                rs += __shfl_xor(rs, off, 32);
            l8[i] = l8[i] * alpha[i] + rs;
            m8[i] = mnew;
        }
#pragma unroll
        for (int t = 0; t < 8; ++t)
#pragma unroll
            for (int i = 0; i < 8; ++i)
                Oacc[t][i] *= alpha[i];
        // stage P through LDS, regather as A-fragment
#pragma unroll
        for (int c = 0; c < 2; ++c)
#pragma unroll
            for (int i = 0; i < 8; ++i)
                Ps[w][i + half * 8][c * 16 + ln] = (_Float16)p[c][i];
        __syncthreads();            // Ps + Vt visible
        FragH pf;
#pragma unroll
        for (int j = 0; j < 8; ++j)
            pf.u[j] = *(const unsigned*)&Ps[w][ln][fragBase(j, half)];
        // O += P * V
#pragma unroll
        for (int nt = 0; nt < 8; ++nt) {
            FragH bv;
            int d = nt * 16 + ln;
#pragma unroll
            for (int j = 0; j < 8; ++j)
                bv.u[j] = *(const unsigned*)&Vt[d][fragBase(j, half)];
            Oacc[nt] = __builtin_amdgcn_wmma_f32_16x16x32_f16(
                false, pf.h, false, bv.h, (short)0, Oacc[nt], false, false);
        }
    }

#pragma unroll
    for (int nt = 0; nt < 8; ++nt)
#pragma unroll
        for (int i = 0; i < 8; ++i) {
            int qg = qb * 128 + w * 16 + i + half * 8;
            float v = Oacc[nt][i] / l8[i];
            O[(size_t)(b * 2048 + qg) * 2048 + h * 128 + nt * 16 + ln] = (_Float16)v;
        }
}

// ---------------- launch ----------------
extern "C" void kernel_launch(void* const* d_in, const int* in_sizes, int n_in,
                              void* d_out, int out_size, void* d_ws, size_t ws_size,
                              hipStream_t stream) {
    const int B = 2, S = 2048, D = 2048;
    const int M = B * S;                  // 4096
    const size_t XE = (size_t)M * D;      // 8,388,608
    const size_t WE = (size_t)D * D;      // 4,194,304

    const float* X  = (const float*)d_in[0];
    const int*  pos = (const int*)d_in[2];
    const float* Wq = (const float*)d_in[3];
    const float* Wk = (const float*)d_in[4];
    const float* Wv = (const float*)d_in[5];
    const float* Wo = (const float*)d_in[6];

    _Float16* Xh  = (_Float16*)d_ws;
    _Float16* Wqh = Xh  + XE;
    _Float16* Wkh = Wqh + WE;
    _Float16* Wvh = Wkh + WE;
    _Float16* Woh = Wvh + WE;
    _Float16* Qh  = Woh + WE;
    _Float16* Kh  = Qh  + XE;
    _Float16* Vh  = Kh  + XE;
    _Float16* Oh  = Vh  + XE;

    cvt_f32_f16<<<4096, 256, 0, stream>>>(X,  Xh,  (int)XE);
    cvt_f32_f16<<<2048, 256, 0, stream>>>(Wq, Wqh, (int)WE);
    cvt_f32_f16<<<2048, 256, 0, stream>>>(Wk, Wkh, (int)WE);
    cvt_f32_f16<<<2048, 256, 0, stream>>>(Wv, Wvh, (int)WE);
    cvt_f32_f16<<<2048, 256, 0, stream>>>(Wo, Woh, (int)WE);

    dim3 gGemm(D / 128, M / 128);
    gemm_f16_wmma<<<gGemm, 256, 0, stream>>>(Xh, Wqh, Qh, nullptr, M, D, D);
    gemm_f16_wmma<<<gGemm, 256, 0, stream>>>(Xh, Wkh, Kh, nullptr, M, D, D);
    gemm_f16_wmma<<<gGemm, 256, 0, stream>>>(Xh, Wvh, Vh, nullptr, M, D, D);

    int ropeTotal = M * 16 * 64;
    rope_kernel<<<(ropeTotal + 255) / 256, 256, 0, stream>>>(Qh, Kh, pos, ropeTotal);

    attn_kernel<<<dim3(S / 128, B * 16), 256, 0, stream>>>(Qh, Kh, Vh, Oh);

    gemm_f16_wmma<<<gGemm, 256, 0, stream>>>(Oh, Woh, nullptr, (float*)d_out, M, D, D);
}
